// Panacea_57234734187179
// MI455X (gfx1250) — matrix operand order repaired and checked by reading, not compiled, not run on hardware
//
#include <hip/hip_runtime.h>
#include <hip/hip_bf16.h>
#include <stdint.h>

// ---------------------------------------------------------------------------
// MI455X (gfx1250) implementation.
//
// Roofline reasoning:
//  * agg / embed node tensors are [50000,256] f32 = 51.2MB each -> L2 (192MB)
//    resident.
//  * Edge aggregation is done as an atomic-free gather over a CSR built once
//    per launch (edge topology is launch-invariant): replaces 409.6M f32
//    atomics/hop with 3.2M int atomics once.  Buckets are sorted by edge id
//    so the f32 accumulation order is fixed -> bit-deterministic output.
//  * GEMMs (~98 GFLOP total) run on the WMMA pipe via the codegen-confirmed
//    v_wmma_f32_16x16x32_f16 (f16 fragments, f32 accumulation).  Gate MLP is
//    fused (GEMM -> relu -> dot w2 -> sigmoid), never materializing [N,512].
// ---------------------------------------------------------------------------

typedef __attribute__((ext_vector_type(16))) _Float16 v16h;
typedef __attribute__((ext_vector_type(8)))  float    v8f;
typedef _Float16 half_t;

#define DIM   256
#define HID   512
#define OUTS  4          // n_hops + 1 slices
#define OUTSTRIDE (OUTS*DIM)

// ---- WMMA fragment loaders -------------------------------------------------
// 16-bit A matrix 16x32 (MxK) layout (matches B): lane reads two contiguous
// 8-half chunks: [k0+hi*16, +8) and [k0+hi*16+8, +8), hi = lane>>4.
__device__ __forceinline__ v16h load_a_frag(const half_t* __restrict__ ldsA,
                                            int strideH, int lane, int k0) {
    int m  = lane & 15;
    int hi = lane >> 4;
    const half_t* p = ldsA + m * strideH;
    v16h a;
    int b0 = k0 + hi * 16;
    int b1 = k0 + hi * 16 + 8;
#pragma unroll
    for (int j = 0; j < 8; ++j) a[j]     = p[b0 + j];
#pragma unroll
    for (int j = 0; j < 8; ++j) a[8 + j] = p[b1 + j];
    return a;
}

// 16-bit B matrix 32x16 (KxN): lanes 0-15 hold K=0..15 of col=lane, lanes
// 16-31 hold K=16..31.  BT is stored [col][K] (f16, transposed) -> one
// contiguous 32B read per lane.
__device__ __forceinline__ v16h load_b_frag(const half_t* __restrict__ BT,
                                            int K, int lane, int k0, int col0) {
    int col = col0 + (lane & 15);
    int hi  = lane >> 4;
    const half_t* p = BT + (size_t)col * K + k0 + hi * 16;
    v16h b;
#pragma unroll
    for (int j = 0; j < 16; ++j) b[j] = p[j];
    return b;
}

__device__ __forceinline__ v8f wmma_f16(v16h a, v16h b, v8f c) {
    return __builtin_amdgcn_wmma_f32_16x16x32_f16(false, a, false, b,
                                                  (short)0, c, false, false);
}

// ---- prep: transpose f32 weight [K][C] -> f16 [C][K] ------------------------
__global__ __launch_bounds__(256) void prep_wT_kernel(
    const float* __restrict__ W, half_t* __restrict__ WT, int K, int C) {
    int i = blockIdx.x * 256 + threadIdx.x;
    if (i >= K * C) return;
    int k = i / C, c = i % C;                 // coalesced read over c
    WT[(size_t)c * K + k] = (half_t)W[i];
}

// ---- out[:,0,:] = embed -----------------------------------------------------
__global__ __launch_bounds__(256) void copy_slice0_kernel(
    const float* __restrict__ embed, float* __restrict__ out, long long total) {
    long long i = (long long)blockIdx.x * 256 + threadIdx.x;
    if (i >= total) return;
    long long n = i >> 8;                     // /DIM
    int d = (int)(i & (DIM - 1));
    out[n * OUTSTRIDE + d] = embed[i];
}

// ============================ CSR construction ==============================
__global__ __launch_bounds__(256) void count_kernel(
    const long long* __restrict__ colIdx, int* __restrict__ count, long long E) {
    long long i = (long long)blockIdx.x * 256 + threadIdx.x;
    if (i >= E) return;
    atomicAdd(&count[(int)colIdx[i]], 1);
}

// single-block exclusive scan of count[N] -> start[N+1]
__global__ __launch_bounds__(1024) void scan_kernel(
    const int* __restrict__ count, int* __restrict__ start, int N) {
    __shared__ int sums[1024];
    int tid = threadIdx.x;
    int chunk = (N + 1023) / 1024;
    int lo = tid * chunk;
    int hi = lo + chunk; if (hi > N) hi = N; if (lo > N) lo = N;
    int s = 0;
    for (int i = lo; i < hi; ++i) s += count[i];
    sums[tid] = s;
    __syncthreads();
    for (int d = 1; d < 1024; d <<= 1) {      // Hillis-Steele inclusive scan
        int v = (tid >= d) ? sums[tid - d] : 0;
        __syncthreads();
        sums[tid] += v;
        __syncthreads();
    }
    int prefix = (tid == 0) ? 0 : sums[tid - 1];
    for (int i = lo; i < hi; ++i) {
        start[i] = prefix;
        prefix += count[i];
    }
    if (tid == 1023) start[N] = prefix;
}

__global__ __launch_bounds__(256) void fill_kernel(
    const long long* __restrict__ colIdx, const int* __restrict__ start,
    int* __restrict__ cursor, int* __restrict__ edgeList, long long E) {
    long long i = (long long)blockIdx.x * 256 + threadIdx.x;
    if (i >= E) return;
    int c = (int)colIdx[i];
    int pos = start[c] + atomicAdd(&cursor[c], 1);
    edgeList[pos] = (int)i;
}

// sort each bucket ascending by edge id -> deterministic accumulation order
__global__ __launch_bounds__(256) void sort_buckets_kernel(
    const int* __restrict__ start, int* __restrict__ edgeList, int N) {
    int n = blockIdx.x * 256 + threadIdx.x;
    if (n >= N) return;
    int s0 = start[n], s1 = start[n + 1];
    for (int i = s0 + 1; i < s1; ++i) {
        int v = edgeList[i];
        int j = i - 1;
        while (j >= s0 && edgeList[j] > v) { edgeList[j + 1] = edgeList[j]; --j; }
        edgeList[j + 1] = v;
    }
}

// ---- atomic-free gather: agg[n,:] = sum_{e in bucket(n)} src[row[e],:]*w[e]
// One block per destination node; thread t owns feature dim t.
__global__ __launch_bounds__(256) void gather_kernel(
    const float* __restrict__ src, long long srcStride,
    const long long* __restrict__ rowIdx, const float* __restrict__ adap,
    const int* __restrict__ start, const int* __restrict__ edgeList,
    float* __restrict__ agg) {
    __shared__ int   sRow[256];
    __shared__ float sW[256];
    int n   = blockIdx.x;
    int tid = threadIdx.x;
    int s0 = start[n], s1 = start[n + 1];
    float acc = 0.f;
    for (int base = s0; base < s1; base += 256) {
        int cnt = s1 - base; if (cnt > 256) cnt = 256;
        if (tid < cnt) {
            int e = edgeList[base + tid];
            sRow[tid] = (int)rowIdx[e];
            sW[tid]   = adap[e];
        }
        __syncthreads();
        for (int i = 0; i < cnt; ++i) {
            acc += src[(size_t)sRow[i] * srcStride + tid] * sW[i];
        }
        __syncthreads();
    }
    agg[(size_t)n * DIM + tid] = acc;
}

// ---- fused gate MLP: cum[n] += sigmoid(relu([agg,embed]@w1 + b1)@w2 + b2) ---
#define GATE_LDA (HID + 8)
__global__ __launch_bounds__(256) void gate_kernel(
    const float* __restrict__ agg, const float* __restrict__ embed,
    const half_t* __restrict__ w1T, const float* __restrict__ b1,
    const float* __restrict__ w2, const float* __restrict__ b2,
    float* __restrict__ cum, int N) {
    __shared__ half_t ldsA[16 * GATE_LDA];
    __shared__ float  ldsGate[16];
    int tid = threadIdx.x;
    int n0  = blockIdx.x * 16;

    for (int idx = tid; idx < 16 * HID; idx += 256) {
        int m = idx >> 9;
        int k = idx & (HID - 1);
        int n = n0 + m; if (n >= N) n = N - 1;   // clamp: keep EXEC full
        float v = (k < DIM) ? agg[(size_t)n * DIM + k]
                            : embed[(size_t)n * DIM + (k - DIM)];
        ldsA[m * GATE_LDA + k] = (half_t)v;
    }
    if (tid < 16) ldsGate[tid] = 0.f;
    __syncthreads();

    int wave = tid >> 5, lane = tid & 31;
    v8f zero = {};
    v8f acc[4];
#pragma unroll
    for (int t = 0; t < 4; ++t) acc[t] = zero;

    for (int k0 = 0; k0 < HID; k0 += 32) {
        v16h a = load_a_frag(ldsA, GATE_LDA, lane, k0);
#pragma unroll
        for (int t = 0; t < 4; ++t) {
            v16h b = load_b_frag(w1T, HID, lane, k0, (wave * 4 + t) * 16);
            acc[t] = wmma_f16(a, b, acc[t]);
        }
    }

    int hi = lane >> 4, colL = lane & 15;
    float partial[8];
#pragma unroll
    for (int r = 0; r < 8; ++r) partial[r] = 0.f;
#pragma unroll
    for (int t = 0; t < 4; ++t) {
        int col = (wave * 4 + t) * 16 + colL;
        float b1v = b1[col], w2v = w2[col];
#pragma unroll
        for (int r = 0; r < 8; ++r) {
            float h = acc[t][r] + b1v;
            h = h > 0.f ? h : 0.f;
            partial[r] += h * w2v;
        }
    }
#pragma unroll
    for (int r = 0; r < 8; ++r) {
        partial[r] += __shfl_xor(partial[r], 8, 32);
        partial[r] += __shfl_xor(partial[r], 4, 32);
        partial[r] += __shfl_xor(partial[r], 2, 32);
        partial[r] += __shfl_xor(partial[r], 1, 32);
    }
    if ((lane & 15) == 0) {
        int mbase = hi * 8;
#pragma unroll
        for (int r = 0; r < 8; ++r) atomicAdd(&ldsGate[mbase + r], partial[r]);
    }
    __syncthreads();
    if (tid < 16 && (n0 + tid) < N) {
        float g = 1.f / (1.f + __expf(-(ldsGate[tid] + b2[0])));
        cum[n0 + tid] += g;
    }
}

// ---- mlp update: out[:,h+1,:] = agg@W + b + agg + (1-cum)*embed + out[:,h,:]
#define MLP_LDA (DIM + 8)
#define MLP_LDF (DIM + 4)
__global__ __launch_bounds__(256) void mlp_kernel(
    const float* __restrict__ agg, const float* __restrict__ embed,
    const half_t* __restrict__ mwT, const float* __restrict__ mb,
    const float* __restrict__ cum, float* __restrict__ out, int hop, int N) {
    __shared__ half_t ldsA[16 * MLP_LDA];
    __shared__ float  ldsF[16 * MLP_LDF];
    __shared__ float  ldsCum[16];
    int tid = threadIdx.x;
    int n0  = blockIdx.x * 16;

    for (int idx = tid; idx < 16 * DIM; idx += 256) {
        int m = idx >> 8;
        int k = idx & (DIM - 1);
        int n = n0 + m; if (n >= N) n = N - 1;
        float v = agg[(size_t)n * DIM + k];
        ldsA[m * MLP_LDA + k] = (half_t)v;
        ldsF[m * MLP_LDF + k] = v;
    }
    if (tid < 16) {
        int n = n0 + tid; if (n >= N) n = N - 1;
        ldsCum[tid] = cum[n];
    }
    __syncthreads();

    int wave = tid >> 5, lane = tid & 31;
    v8f zero = {};
    v8f acc[2];
    acc[0] = zero; acc[1] = zero;

    for (int k0 = 0; k0 < DIM; k0 += 32) {
        v16h a = load_a_frag(ldsA, MLP_LDA, lane, k0);
#pragma unroll
        for (int t = 0; t < 2; ++t) {
            v16h b = load_b_frag(mwT, DIM, lane, k0, (wave * 2 + t) * 16);
            acc[t] = wmma_f16(a, b, acc[t]);
        }
    }

    int hi = lane >> 4, colL = lane & 15;
#pragma unroll
    for (int t = 0; t < 2; ++t) {
        int col = (wave * 2 + t) * 16 + colL;
        float mbv = mb[col];
#pragma unroll
        for (int r = 0; r < 8; ++r) {
            int m = hi * 8 + r;
            long long n = n0 + m;
            if (n >= N) continue;             // after WMMA: guards are safe
            float aggv = ldsF[m * MLP_LDF + col];
            float ev   = embed[(size_t)n * DIM + col];
            float prev = out[n * OUTSTRIDE + (size_t)hop * DIM + col];
            float y = acc[t][r] + mbv + aggv + (1.f - ldsCum[m]) * ev + prev;
            out[n * OUTSTRIDE + (size_t)(hop + 1) * DIM + col] = y;
        }
    }
}

// ---------------------------------------------------------------------------
extern "C" void kernel_launch(void* const* d_in, const int* in_sizes, int n_in,
                              void* d_out, int out_size, void* d_ws, size_t ws_size,
                              hipStream_t stream) {
    const float*     embed = (const float*)d_in[0];
    const float*     adap  = (const float*)d_in[1];
    const float*     mlp_w = (const float*)d_in[2];
    const float*     mlp_b = (const float*)d_in[3];
    const float*     tm_w1 = (const float*)d_in[4];
    const float*     tm_b1 = (const float*)d_in[5];
    const float*     tm_w2 = (const float*)d_in[6];
    const float*     tm_b2 = (const float*)d_in[7];
    const long long* eidx  = (const long long*)d_in[8];

    const int       N = in_sizes[0] / DIM;
    const long long E = in_sizes[1];
    const long long* rowIdx = eidx;
    const long long* colIdx = eidx + E;

    float* out = (float*)d_out;

    // workspace layout (256B aligned)
    auto align256 = [](size_t x) { return (x + 255) & ~(size_t)255; };
    uint8_t* ws   = (uint8_t*)d_ws;
    size_t   off  = 0;
    float*  aggB  = (float*)(ws + off);  off = align256(off + (size_t)N * DIM * 4);
    float*  cum   = (float*)(ws + off);  off = align256(off + (size_t)N * 4);
    half_t* w1T   = (half_t*)(ws + off); off = align256(off + (size_t)2 * DIM * HID * 2);
    half_t* mwT   = (half_t*)(ws + off); off = align256(off + (size_t)DIM * DIM * 2);
    int*    count = (int*)(ws + off);    off = align256(off + (size_t)N * 4);
    int*    start = (int*)(ws + off);    off = align256(off + (size_t)(N + 1) * 4);
    int*    elist = (int*)(ws + off);    off = align256(off + (size_t)E * 4);
    (void)ws_size;

    const unsigned edgeBlocks = (unsigned)((E + 255) / 256);
    const int nodeBlocks16 = (N + 15) / 16;
    const int nodeBlocks256 = (N + 255) / 256;

    // ---- one-time per launch: CSR by destination, deterministic order ----
    hipMemsetAsync(count, 0, (size_t)N * 4, stream);
    count_kernel<<<edgeBlocks, 256, 0, stream>>>(colIdx, count, E);
    scan_kernel<<<1, 1024, 0, stream>>>(count, start, N);
    hipMemsetAsync(count, 0, (size_t)N * 4, stream);       // reuse as cursor
    fill_kernel<<<edgeBlocks, 256, 0, stream>>>(colIdx, start, count, elist, E);
    sort_buckets_kernel<<<nodeBlocks256, 256, 0, stream>>>(start, elist, N);

    // ---- weights -> f16 transposed; out slice 0; cum = 0 ----
    hipMemsetAsync(cum, 0, (size_t)N * 4, stream);
    prep_wT_kernel<<<(2 * DIM * HID + 255) / 256, 256, 0, stream>>>(tm_w1, w1T, 2 * DIM, HID);
    prep_wT_kernel<<<(DIM * DIM + 255) / 256, 256, 0, stream>>>(mlp_w, mwT, DIM, DIM);
    long long tot0 = (long long)N * DIM;
    copy_slice0_kernel<<<(unsigned)((tot0 + 255) / 256), 256, 0, stream>>>(embed, out, tot0);

    // ---- hops ----
    for (int hop = 0; hop < 3; ++hop) {
        const float* src = (hop == 0) ? embed : (out + (size_t)hop * DIM);
        long long srcStride = (hop == 0) ? DIM : OUTSTRIDE;

        gather_kernel<<<N, 256, 0, stream>>>(src, srcStride, rowIdx, adap,
                                             start, elist, aggB);

        gate_kernel<<<nodeBlocks16, 256, 0, stream>>>(aggB, embed, w1T, tm_b1,
                                                      tm_w2, tm_b2, cum, N);

        mlp_kernel<<<nodeBlocks16, 256, 0, stream>>>(aggB, embed, mwT, mlp_b,
                                                     cum, out, hop, N);
    }
}
